// DifferentialAttention_66889820668289
// MI455X (gfx1250) — matrix operand order repaired and checked by reading, not compiled, not run on hardware
//
#include <hip/hip_runtime.h>
#include <hip/hip_bf16.h>

// ---------------------------------------------------------------------------
// Differential attention, MI455X (gfx1250): wave32, bf16 WMMA, flash-style,
// double-buffered TDM (tensor_load_to_lds) staging of K/V tiles.
// ---------------------------------------------------------------------------

typedef __bf16 bf16_t;
typedef __attribute__((ext_vector_type(8)))  __bf16 v8bf;
typedef __attribute__((ext_vector_type(16))) __bf16 v16bf;
typedef __attribute__((ext_vector_type(8)))  float  v8f;
typedef __attribute__((ext_vector_type(4)))  unsigned int u32x4;
typedef __attribute__((ext_vector_type(8)))  int  i32x8;
typedef __attribute__((ext_vector_type(4)))  int  i32x4;

#define TSEQ   2048
#define DMODEL 1024
#define NHEAD  16
#define NKVH   8
#define HDIM   64

static constexpr float LAMBDA_INIT_F = 0.35550907f;   // 0.8 - 0.6*exp(-0.3)

// -------------------------------- WMMA helpers -----------------------------

__device__ __forceinline__ v8f wmma_bf16(v16bf a, v16bf b, v8f c) {
  return __builtin_amdgcn_wmma_f32_16x16x32_bf16(
      false, a, false, b, (short)0, c, false, false);
}

// A fragment (16x32), source row-major, leading dim ld.
__device__ __forceinline__ v16bf load_a_frag(const bf16_t* __restrict__ p0, int ld) {
  const int lane = threadIdx.x & 31;
  const bf16_t* p = p0 + (size_t)(lane & 15) * ld + ((lane >> 4) << 3);
  v8bf lo = *(const v8bf*)(p);
  v8bf hi = *(const v8bf*)(p + 16);
  return __builtin_shufflevector(lo, hi, 0,1,2,3,4,5,6,7,8,9,10,11,12,13,14,15);
}

// B fragment (K=32 x N=16); source is B^T stored row-major (N x K), ld = K pitch.
__device__ __forceinline__ v16bf load_b_frag(const bf16_t* __restrict__ p0, int ld) {
  const int lane = threadIdx.x & 31;
  const bf16_t* p = p0 + (size_t)(lane & 15) * ld + ((lane >> 4) << 3);
  v8bf lo = *(const v8bf*)(p);
  v8bf hi = *(const v8bf*)(p + 16);
  return __builtin_shufflevector(lo, hi, 0,1,2,3,4,5,6,7,8,9,10,11,12,13,14,15);
}

// -------------------------- TDM tile load ----------------------------------
// 2D tile (tile_d1 rows x tile_d0 elements, bf16) from a row-major tensor with
// row pitch stride0 (elements) into LDS at lds_off (packed row-major).
// D# layout per CDNA5 ISA ch.8 (group0: count/lds/global/type; group1: dims).

__device__ __forceinline__ unsigned lds_offset_of(const void* p) {
  // generic pointer to LDS: addr[31:0] is the group-segment byte offset
  return (unsigned)(unsigned long long)p;
}

__device__ __forceinline__ void tdm_load_tile_2d(
    unsigned lds_off, const void* gptr,
    unsigned tile_d0, unsigned tile_d1,
    unsigned long long tens_d0, unsigned long long tens_d1,
    unsigned long long stride0) {
  const unsigned long long ga = (unsigned long long)gptr;
  u32x4 g0;
  g0.x = 1u;                                             // count=1 (valid), user mode
  g0.y = lds_off;                                        // lds_addr
  g0.z = (unsigned)ga;                                   // global_addr[31:0]
  g0.w = ((unsigned)(ga >> 32) & 0x01FFFFFFu) | (2u << 30); // addr[56:32] | type=2

  // group1 bitfields (256b): data_size@16 (1->2B), tensor_dim0@48(32b),
  // tensor_dim1@80(32b), tile_dim0@112(16b), tile_dim1@128(16b),
  // tensor_dim0_stride@160(48b), tensor_dim1_stride@208(48b)=0
  const unsigned long long q0 = (1ull << 16) | ((tens_d0 & 0xFFFFull) << 48);
  const unsigned long long q1 = (tens_d0 >> 16) | (tens_d1 << 16)
                              | ((unsigned long long)(tile_d0 & 0xFFFFu) << 48);
  const unsigned long long q2 = (unsigned long long)(tile_d1 & 0xFFFFu)
                              | ((stride0 & 0xFFFFFFFFull) << 32);
  const unsigned long long q3 = (stride0 >> 32) & 0xFFFFull;
  i32x8 g1;
  g1[0] = (int)q0; g1[1] = (int)(q0 >> 32);
  g1[2] = (int)q1; g1[3] = (int)(q1 >> 32);
  g1[4] = (int)q2; g1[5] = (int)(q2 >> 32);
  g1[6] = (int)q3; g1[7] = (int)(q3 >> 32);
  const i32x4 z4 = {0, 0, 0, 0};
  const i32x8 z8 = {0, 0, 0, 0, 0, 0, 0, 0};
  __builtin_amdgcn_tensor_load_to_lds(g0, g1, z4, z4, z8, 0);
}

// ------------------------------ elementwise --------------------------------

__global__ void cvt_f32_bf16(const float* __restrict__ src,
                             bf16_t* __restrict__ dst, int n) {
  int i = blockIdx.x * blockDim.x + threadIdx.x;
  if (i < n) dst[i] = (bf16_t)src[i];
}

// ---------------------------------- GEMM -----------------------------------
// C(MxN f32) = A(MxK bf16) * B(NxK bf16)^T. One wave = 16x64 C tile
// (A fragment reused across 4 N-subtiles); block = 4 waves along M.
__global__ __launch_bounds__(128) void gemm_bf16_nt(
    float* __restrict__ C, const bf16_t* __restrict__ A,
    const bf16_t* __restrict__ B, int M, int N, int K) {
  const int wave = threadIdx.x >> 5;
  const int lane = threadIdx.x & 31;
  const int n0 = blockIdx.x * 64;
  const int m0 = (blockIdx.y * 4 + wave) * 16;

  const bf16_t* arow = A + (size_t)m0 * K;
  v8f acc[4];
  #pragma unroll
  for (int j = 0; j < 4; ++j) acc[j] = (v8f){0.f,0.f,0.f,0.f,0.f,0.f,0.f,0.f};

  for (int k0 = 0; k0 < K; k0 += 32) {
    v16bf af = load_a_frag(arow + k0, K);
    #pragma unroll
    for (int j = 0; j < 4; ++j) {
      v16bf bfr = load_b_frag(B + (size_t)(n0 + j * 16) * K + k0, K);
      acc[j] = wmma_bf16(af, bfr, acc[j]);
    }
  }
  const int col = lane & 15;
  const int rb  = (lane >> 4) * 8;
  #pragma unroll
  for (int j = 0; j < 4; ++j)
    #pragma unroll
    for (int i = 0; i < 8; ++i)
      C[(size_t)(m0 + rb + i) * N + n0 + j * 16 + col] = acc[j][i];
}

// --------------------------- RoPE + repack ---------------------------------

__global__ void rope_pack_q(const float* __restrict__ qf,
                            const float* __restrict__ fc,
                            const float* __restrict__ fs,
                            bf16_t* __restrict__ qb) {
  int idx = blockIdx.x * blockDim.x + threadIdx.x;
  int j    =  idx        & 31;
  int comp = (idx >> 5)  & 1;
  int h    = (idx >> 6)  & 15;
  int t    =  idx >> 10;
  const float* src = qf + (size_t)t * (NHEAD * 2 * HDIM) + h * (2 * HDIM) + comp * HDIM + 2 * j;
  float r = src[0], im = src[1];
  float c = fc[t * (HDIM / 2) + j], s = fs[t * (HDIM / 2) + j];
  bf16_t* dst = qb + (((size_t)h * 2 + comp) * TSEQ + t) * HDIM + 2 * j;
  dst[0] = (bf16_t)(r * c - im * s);
  dst[1] = (bf16_t)(r * s + im * c);
}

__global__ void rope_pack_k(const float* __restrict__ kf,
                            const float* __restrict__ fc,
                            const float* __restrict__ fs,
                            bf16_t* __restrict__ kb) {
  int idx = blockIdx.x * blockDim.x + threadIdx.x;
  int j    =  idx        & 31;
  int comp = (idx >> 5)  & 1;
  int kv   = (idx >> 6)  & 7;
  int t    =  idx >> 9;
  const float* src = kf + (size_t)t * (NKVH * 2 * HDIM) + kv * (2 * HDIM) + comp * HDIM + 2 * j;
  float r = src[0], im = src[1];
  float c = fc[t * (HDIM / 2) + j], s = fs[t * (HDIM / 2) + j];
  bf16_t* dst = kb + (((size_t)kv * 2 + comp) * TSEQ + t) * HDIM + 2 * j;
  dst[0] = (bf16_t)(r * c - im * s);
  dst[1] = (bf16_t)(r * s + im * c);
}

__global__ void pack_vT(const float* __restrict__ vf, bf16_t* __restrict__ vT) {
  int idx = blockIdx.x * blockDim.x + threadIdx.x;
  int t  =  idx         & (TSEQ - 1);
  int d  = (idx >> 11)  & 63;
  int kv =  idx >> 17;
  vT[idx] = (bf16_t)vf[(size_t)t * (NKVH * HDIM) + kv * HDIM + d];
}

__global__ void lambda_kernel(const float* __restrict__ lq1, const float* __restrict__ lk1,
                              const float* __restrict__ lq2, const float* __restrict__ lk2,
                              float* __restrict__ lam) {
  if (threadIdx.x == 0) {
    float s1 = 0.f, s2 = 0.f;
    for (int i = 0; i < HDIM; ++i) { s1 += lq1[i] * lk1[i]; s2 += lq2[i] * lk2[i]; }
    *lam = __expf(s1) - __expf(s2) + LAMBDA_INIT_F;
  }
}

// ------------------------- flash differential attention --------------------
// K tiles staged in LDS as [32 keys][64 hd], V tile as [64 hd][32 keys],
// double-buffered so the TDM fills tile i+1 while tile i is consumed.

__device__ __forceinline__ void attn_step(
    v16bf qf0, v16bf qf1,
    const bf16_t* __restrict__ Klds,  // [32][64] staged tile
    const bf16_t* __restrict__ Vlds,  // [64][32] staged tile
    int s0, int q0, float scale,
    float m[8], float l[8], v8f o[4], bf16_t* pb) {
  const int lane = threadIdx.x & 31;
  const int rb   = (lane >> 4) * 8;
  const int col  = lane & 15;

  v8f sacc[2];
  #pragma unroll
  for (int sub = 0; sub < 2; ++sub) {
    const bf16_t* kbase = Klds + (size_t)(sub * 16) * HDIM;
    v16bf k0 = load_b_frag(kbase, HDIM);
    v16bf k1 = load_b_frag(kbase + 32, HDIM);
    v8f acc = {0.f,0.f,0.f,0.f,0.f,0.f,0.f,0.f};
    acc = wmma_bf16(qf0, k0, acc);
    acc = wmma_bf16(qf1, k1, acc);
    sacc[sub] = acc;
  }

  float rmax[8];
  #pragma unroll
  for (int i = 0; i < 8; ++i) {
    const int r = q0 + rb + i;
    float a = (s0 + col      <= r) ? sacc[0][i] * scale : -__builtin_inff();
    float b = (s0 + 16 + col <= r) ? sacc[1][i] * scale : -__builtin_inff();
    sacc[0][i] = a; sacc[1][i] = b;
    rmax[i] = fmaxf(a, b);
  }
  #pragma unroll
  for (int i = 0; i < 8; ++i) {
    float v = rmax[i];
    v = fmaxf(v, __shfl_xor(v, 1, 32));
    v = fmaxf(v, __shfl_xor(v, 2, 32));
    v = fmaxf(v, __shfl_xor(v, 4, 32));
    v = fmaxf(v, __shfl_xor(v, 8, 32));
    rmax[i] = v;
  }

  #pragma unroll
  for (int i = 0; i < 8; ++i) {
    const float mn = fmaxf(m[i], rmax[i]);
    const float p0 = __expf(sacc[0][i] - mn);
    const float p1 = __expf(sacc[1][i] - mn);
    pb[(rb + i) * 32 + col]      = (bf16_t)p0;
    pb[(rb + i) * 32 + 16 + col] = (bf16_t)p1;
    float ps = p0 + p1;
    ps += __shfl_xor(ps, 1, 32);
    ps += __shfl_xor(ps, 2, 32);
    ps += __shfl_xor(ps, 4, 32);
    ps += __shfl_xor(ps, 8, 32);
    const float corr = __expf(m[i] - mn);
    m[i] = mn;
    l[i] = l[i] * corr + ps;
    #pragma unroll
    for (int j = 0; j < 4; ++j) o[j][i] *= corr;
  }

  v16bf pf = load_a_frag(pb, 32);
  #pragma unroll
  for (int j = 0; j < 4; ++j) {
    v16bf vfrag = load_b_frag(Vlds + (size_t)(j * 16) * 32, 32);
    o[j] = wmma_bf16(pf, vfrag, o[j]);
  }
}

__global__ __launch_bounds__(128) void diff_attn_kernel(
    const bf16_t* __restrict__ qb,   // [NH][2][T][HD]
    const bf16_t* __restrict__ kb,   // [NKV][2][T][HD]
    const bf16_t* __restrict__ vT,   // [NKV][HD][T]
    const float* __restrict__ lam_p,
    float* __restrict__ yf) {        // [T][NH*HD]
  __shared__ __align__(16) bf16_t kstage[4][2][2][32 * 64]; // [wave][buf][map]
  __shared__ __align__(16) bf16_t vstage[4][2][64 * 32];    // [wave][buf]
  __shared__ __align__(16) bf16_t pbuf[4][2][16 * 32];      // P staging
  const int wave = threadIdx.x >> 5;
  const int lane = threadIdx.x & 31;
  const int h    = blockIdx.x;
  const int q0   = (blockIdx.y * 4 + wave) * 16;
  const int kv   = h >> 1;                  // N_REP = 2
  const float lam = *lam_p;
  const float scale = 0.125f;               // 1/sqrt(64)

  const bf16_t* Q1 = qb + ((size_t)(h * 2 + 0) * TSEQ) * HDIM;
  const bf16_t* Q2 = qb + ((size_t)(h * 2 + 1) * TSEQ) * HDIM;
  const bf16_t* K1 = kb + ((size_t)(kv * 2 + 0) * TSEQ) * HDIM;
  const bf16_t* K2 = kb + ((size_t)(kv * 2 + 1) * TSEQ) * HDIM;
  const bf16_t* V  = vT + (size_t)kv * HDIM * TSEQ;

  // issue one 32-key tile set (K1, K2, V) into buffer `buf`
  auto issue_tiles = [&](int s0, int buf) {
    tdm_load_tile_2d(lds_offset_of(&kstage[wave][buf][0][0]),
                     K1 + (size_t)s0 * HDIM, HDIM, 32, HDIM, TSEQ, HDIM);
    tdm_load_tile_2d(lds_offset_of(&kstage[wave][buf][1][0]),
                     K2 + (size_t)s0 * HDIM, HDIM, 32, HDIM, TSEQ, HDIM);
    tdm_load_tile_2d(lds_offset_of(&vstage[wave][buf][0]),
                     V + s0, 32, HDIM, TSEQ, HDIM, TSEQ);
  };

  v16bf q1a = load_a_frag(Q1 + (size_t)q0 * HDIM,      HDIM);
  v16bf q1b = load_a_frag(Q1 + (size_t)q0 * HDIM + 32, HDIM);
  v16bf q2a = load_a_frag(Q2 + (size_t)q0 * HDIM,      HDIM);
  v16bf q2b = load_a_frag(Q2 + (size_t)q0 * HDIM + 32, HDIM);

  float m1[8], l1[8], m2[8], l2[8];
  v8f o1[4], o2[4];
  #pragma unroll
  for (int i = 0; i < 8; ++i) {
    m1[i] = -__builtin_inff(); m2[i] = -__builtin_inff();
    l1[i] = 0.f; l2[i] = 0.f;
  }
  #pragma unroll
  for (int j = 0; j < 4; ++j)
    #pragma unroll
    for (int i = 0; i < 8; ++i) { o1[j][i] = 0.f; o2[j][i] = 0.f; }

  const int ntiles = (q0 + 15) / 32 + 1;   // causal: keys [0, q0+16)
  issue_tiles(0, 0);
  for (int it = 0; it < ntiles; ++it) {
    const int s0  = it * 32;
    const int buf = it & 1;
    if (it + 1 < ntiles) {
      issue_tiles(s0 + 32, buf ^ 1);       // prefetch next set into other buffer
      asm volatile("" ::: "memory");
      __builtin_amdgcn_s_wait_tensorcnt(3); // in-order: current set (3 ops) done
    } else {
      asm volatile("" ::: "memory");
      __builtin_amdgcn_s_wait_tensorcnt(0);
    }
    asm volatile("" ::: "memory");

    attn_step(q1a, q1b, &kstage[wave][buf][0][0], &vstage[wave][buf][0],
              s0, q0, scale, m1, l1, o1, &pbuf[wave][0][0]);
    attn_step(q2a, q2b, &kstage[wave][buf][1][0], &vstage[wave][buf][0],
              s0, q0, scale, m2, l2, o2, &pbuf[wave][1][0]);
  }

  const int rb  = (lane >> 4) * 8;
  const int col = lane & 15;
  #pragma unroll
  for (int j = 0; j < 4; ++j)
    #pragma unroll
    for (int i = 0; i < 8; ++i) {
      const int r = q0 + rb + i;
      const float y = o1[j][i] / l1[i] - lam * (o2[j][i] / l2[i]);
      yf[(size_t)r * (NHEAD * HDIM) + h * HDIM + j * 16 + col] = y;
    }
}

// ------------------------- per-token normalization -------------------------

__global__ __launch_bounds__(256) void norm_kernel(const float* __restrict__ yf,
                                                   bf16_t* __restrict__ ynb) {
  __shared__ float ssum[8], ssq[8];
  const int t = blockIdx.x;
  const float* row = yf + (size_t)t * DMODEL;
  float s = 0.f, sq = 0.f;
  for (int i = threadIdx.x; i < DMODEL; i += 256) {
    float v = row[i]; s += v; sq += v * v;
  }
  for (int off = 1; off < 32; off <<= 1) {
    s  += __shfl_xor(s,  off, 32);
    sq += __shfl_xor(sq, off, 32);
  }
  const int wave = threadIdx.x >> 5, lane = threadIdx.x & 31;
  if (lane == 0) { ssum[wave] = s; ssq[wave] = sq; }
  __syncthreads();
  float ts = 0.f, tq = 0.f;
  #pragma unroll
  for (int w = 0; w < 8; ++w) { ts += ssum[w]; tq += ssq[w]; }
  const float mean = ts / DMODEL;
  const float var  = tq / DMODEL - mean * mean;
  const float inv  = rsqrtf(var + 1e-5f) * (1.0f - LAMBDA_INIT_F);
  for (int i = threadIdx.x; i < DMODEL; i += 256)
    ynb[(size_t)t * DMODEL + i] = (bf16_t)((row[i] - mean) * inv);
}

// -------------------------------- launch -----------------------------------
static constexpr size_t SZ_XB  = (size_t)TSEQ * DMODEL * 2;
static constexpr size_t SZ_WQB = (size_t)2 * NHEAD * HDIM * DMODEL * 2;
static constexpr size_t SZ_WKB = (size_t)2 * NKVH * HDIM * DMODEL * 2;
static constexpr size_t SZ_WVB = (size_t)NKVH * HDIM * DMODEL * 2;
static constexpr size_t SZ_WOB = (size_t)DMODEL * DMODEL * 2;
static constexpr size_t SZ_QF  = (size_t)TSEQ * 2 * NHEAD * HDIM * 4;  // reused as yf
static constexpr size_t SZ_KF  = (size_t)TSEQ * 2 * NKVH * HDIM * 4;   // reused as ynb
static constexpr size_t SZ_VF  = (size_t)TSEQ * NKVH * HDIM * 4;
static constexpr size_t SZ_QB  = (size_t)NHEAD * 2 * TSEQ * HDIM * 2;
static constexpr size_t SZ_KB  = (size_t)NKVH * 2 * TSEQ * HDIM * 2;
static constexpr size_t SZ_VT  = (size_t)NKVH * HDIM * TSEQ * 2;

static constexpr size_t OFF_XB  = 0;
static constexpr size_t OFF_WQB = OFF_XB  + SZ_XB;
static constexpr size_t OFF_WKB = OFF_WQB + SZ_WQB;
static constexpr size_t OFF_WVB = OFF_WKB + SZ_WKB;
static constexpr size_t OFF_WOB = OFF_WVB + SZ_WVB;
static constexpr size_t OFF_QF  = OFF_WOB + SZ_WOB;
static constexpr size_t OFF_KF  = OFF_QF  + SZ_QF;
static constexpr size_t OFF_VF  = OFF_KF  + SZ_KF;
static constexpr size_t OFF_QB  = OFF_VF  + SZ_VF;
static constexpr size_t OFF_KB  = OFF_QB  + SZ_QB;
static constexpr size_t OFF_VT  = OFF_KB  + SZ_KB;
static constexpr size_t OFF_LAM = OFF_VT  + SZ_VT;

extern "C" void kernel_launch(void* const* d_in, const int* in_sizes, int n_in,
                              void* d_out, int out_size, void* d_ws, size_t ws_size,
                              hipStream_t stream) {
  (void)in_sizes; (void)n_in; (void)out_size; (void)ws_size;
  const float* x   = (const float*)d_in[0];
  const float* fc  = (const float*)d_in[1];
  const float* fs  = (const float*)d_in[2];
  const float* wq  = (const float*)d_in[3];
  const float* wk  = (const float*)d_in[4];
  const float* wv  = (const float*)d_in[5];
  const float* wo  = (const float*)d_in[6];
  const float* lq1 = (const float*)d_in[7];
  const float* lk1 = (const float*)d_in[8];
  const float* lq2 = (const float*)d_in[9];
  const float* lk2 = (const float*)d_in[10];
  float* out = (float*)d_out;
  char*  ws  = (char*)d_ws;

  bf16_t* xb   = (bf16_t*)(ws + OFF_XB);
  bf16_t* wqb  = (bf16_t*)(ws + OFF_WQB);
  bf16_t* wkb  = (bf16_t*)(ws + OFF_WKB);
  bf16_t* wvb  = (bf16_t*)(ws + OFF_WVB);
  bf16_t* wob  = (bf16_t*)(ws + OFF_WOB);
  float*  qf   = (float*)(ws + OFF_QF);
  float*  kf   = (float*)(ws + OFF_KF);
  float*  vf   = (float*)(ws + OFF_VF);
  bf16_t* qbuf = (bf16_t*)(ws + OFF_QB);
  bf16_t* kbuf = (bf16_t*)(ws + OFF_KB);
  bf16_t* vTb  = (bf16_t*)(ws + OFF_VT);
  float*  lam  = (float*)(ws + OFF_LAM);
  float*  yf   = (float*)(ws + OFF_QF);   // reuse
  bf16_t* ynb  = (bf16_t*)(ws + OFF_KF);  // reuse

  const int CT = 256;
  cvt_f32_bf16<<<(TSEQ * DMODEL) / CT, CT, 0, stream>>>(x, xb, TSEQ * DMODEL);
  cvt_f32_bf16<<<(2 * NHEAD * HDIM * DMODEL) / CT, CT, 0, stream>>>(wq, wqb, 2 * NHEAD * HDIM * DMODEL);
  cvt_f32_bf16<<<(2 * NKVH * HDIM * DMODEL) / CT, CT, 0, stream>>>(wk, wkb, 2 * NKVH * HDIM * DMODEL);
  cvt_f32_bf16<<<(NKVH * HDIM * DMODEL) / CT, CT, 0, stream>>>(wv, wvb, NKVH * HDIM * DMODEL);
  cvt_f32_bf16<<<(DMODEL * DMODEL) / CT, CT, 0, stream>>>(wo, wob, DMODEL * DMODEL);

  dim3 gq((2 * NHEAD * HDIM) / 64, TSEQ / 64);
  gemm_bf16_nt<<<gq, 128, 0, stream>>>(qf, xb, wqb, TSEQ, 2 * NHEAD * HDIM, DMODEL);
  dim3 gk((2 * NKVH * HDIM) / 64, TSEQ / 64);
  gemm_bf16_nt<<<gk, 128, 0, stream>>>(kf, xb, wkb, TSEQ, 2 * NKVH * HDIM, DMODEL);
  dim3 gv((NKVH * HDIM) / 64, TSEQ / 64);
  gemm_bf16_nt<<<gv, 128, 0, stream>>>(vf, xb, wvb, TSEQ, NKVH * HDIM, DMODEL);

  rope_pack_q<<<(TSEQ * NHEAD * 2 * (HDIM / 2)) / CT, CT, 0, stream>>>(qf, fc, fs, qbuf);
  rope_pack_k<<<(TSEQ * NKVH * 2 * (HDIM / 2)) / CT, CT, 0, stream>>>(kf, fc, fs, kbuf);
  pack_vT<<<(NKVH * HDIM * TSEQ) / CT, CT, 0, stream>>>(vf, vTb);
  lambda_kernel<<<1, 32, 0, stream>>>(lq1, lk1, lq2, lk2, lam);

  dim3 ga(NHEAD, TSEQ / 64);
  diff_attn_kernel<<<ga, 128, 0, stream>>>(qbuf, kbuf, vTb, lam, yf);

  norm_kernel<<<TSEQ, 256, 0, stream>>>(yf, ynb);

  dim3 go(DMODEL / 64, TSEQ / 64);
  gemm_bf16_nt<<<go, 128, 0, stream>>>(out, ynb, wob, TSEQ, DMODEL, DMODEL);
}
